// CHASEModel_66219805770060
// MI455X (gfx1250) — compile-verified, matrix operand order checked by reading, not run on hardware
//
#include <hip/hip_runtime.h>
#include <hip/hip_bf16.h>
#include <math.h>

typedef __attribute__((ext_vector_type(16))) _Float16 v16h;
typedef __attribute__((ext_vector_type(8)))  float    v8f;

__device__ __forceinline__ float fast_rcp(float x) { return __builtin_amdgcn_rcpf(x); }
__device__ __forceinline__ float fast_sigmoid(float x) {
  return fast_rcp(1.0f + __expf(-x));
}

// Load one 16-lane-per-half WMMA fragment (16 halfs) from an f32 row.
// Layout (ISA 7.12.2, 16-bit A/B): lanes 0-15 hold K=kb+0..7 in V0..V3 and
// K=kb+16..23 in V4..V7; lanes 16-31 hold K=kb+8..15 / +24..31 (kb pre-offset).
__device__ __forceinline__ v16h load_frag_f32(const float* __restrict__ row, int kb)
{
  float4 f0 = *(const float4*)(row + kb);
  float4 f1 = *(const float4*)(row + kb + 4);
  float4 f2 = *(const float4*)(row + kb + 16);
  float4 f3 = *(const float4*)(row + kb + 20);
  v16h r;
  r[0]=(_Float16)f0.x;  r[1]=(_Float16)f0.y;  r[2]=(_Float16)f0.z;  r[3]=(_Float16)f0.w;
  r[4]=(_Float16)f1.x;  r[5]=(_Float16)f1.y;  r[6]=(_Float16)f1.z;  r[7]=(_Float16)f1.w;
  r[8]=(_Float16)f2.x;  r[9]=(_Float16)f2.y;  r[10]=(_Float16)f2.z; r[11]=(_Float16)f2.w;
  r[12]=(_Float16)f3.x; r[13]=(_Float16)f3.y; r[14]=(_Float16)f3.z; r[15]=(_Float16)f3.w;
  return r;
}

// ---------------------------------------------------------------------------
// WMMA GEMM:  C[M,N] = act( A[M,K] @ W[N,K]^T + bias[N] )
// One wave computes a 32x32 C tile: 2 A-frags x 2 B-frags -> 4 WMMAs per
// K-step (each operand reused twice), K-loop unrolled 2x -> 8 static
// v_wmma_f32_16x16x32_f16. act: 0 = none, 1 = SiLU.
// Requires M % 32 == 0, N % 32 == 0, K % 32 == 0 (true for every GEMM here).
// ---------------------------------------------------------------------------
__global__ __launch_bounds__(256) void gemm_wmma_f16(
    const float* __restrict__ A, const float* __restrict__ W,
    const float* __restrict__ bias, float* __restrict__ C,
    int M, int N, int K, int act)
{
  const int lane = threadIdx.x & 31;
  const int wave = threadIdx.x >> 5;
  const int tilesN = N >> 5;                       // 32-wide supertiles in N
  const int tile = blockIdx.x * 8 + wave;          // 8 waves / block
  if (tile >= (M >> 5) * tilesN) return;           // wave-uniform exit

  const int tm = (tile / tilesN) << 5;
  const int tn = (tile % tilesN) << 5;
  const int half = lane >> 4;                      // 0: lanes 0-15, 1: lanes 16-31
  const int l16  = lane & 15;

  const float* Arow0 = A + (size_t)(tm + l16) * K;       // A rows tm..tm+15
  const float* Arow1 = A + (size_t)(tm + 16 + l16) * K;  // A rows tm+16..tm+31
  const float* Wrow0 = W + (size_t)(tn + l16) * K;       // B cols tn..tn+15
  const float* Wrow1 = W + (size_t)(tn + 16 + l16) * K;  // B cols tn+16..tn+31

  v8f acc00 = {}, acc01 = {}, acc10 = {}, acc11 = {};
  #pragma unroll 2
  for (int k0 = 0; k0 < K; k0 += 32) {
    const int kb = k0 + half * 8;
    __builtin_prefetch(Arow0 + kb + 32, 0, 3);     // WGP-scope, next K-step
    __builtin_prefetch(Arow1 + kb + 32, 0, 3);
    __builtin_prefetch(Wrow0 + kb + 32, 0, 3);
    __builtin_prefetch(Wrow1 + kb + 32, 0, 3);
    v16h a0 = load_frag_f32(Arow0, kb);
    v16h a1 = load_frag_f32(Arow1, kb);
    v16h b0 = load_frag_f32(Wrow0, kb);
    v16h b1 = load_frag_f32(Wrow1, kb);
    acc00 = __builtin_amdgcn_wmma_f32_16x16x32_f16(false, a0, false, b0,
                                                   (short)0, acc00, false, false);
    acc01 = __builtin_amdgcn_wmma_f32_16x16x32_f16(false, a0, false, b1,
                                                   (short)0, acc01, false, false);
    acc10 = __builtin_amdgcn_wmma_f32_16x16x32_f16(false, a1, false, b0,
                                                   (short)0, acc10, false, false);
    acc11 = __builtin_amdgcn_wmma_f32_16x16x32_f16(false, a1, false, b1,
                                                   (short)0, acc11, false, false);
  }

  // C layout: VGPR r -> (M = tmX + r + half*8, N = tnX + l16)
  const int n0 = tn + l16;
  const int n1 = n0 + 16;
  const float bn0 = bias ? bias[n0] : 0.0f;
  const float bn1 = bias ? bias[n1] : 0.0f;
  #pragma unroll
  for (int r = 0; r < 8; ++r) {
    const int m0 = tm + r + half * 8;
    const int m1 = m0 + 16;
    float v00 = acc00[r] + bn0;
    float v01 = acc01[r] + bn1;
    float v10 = acc10[r] + bn0;
    float v11 = acc11[r] + bn1;
    if (act == 1) {
      v00 *= fast_sigmoid(v00); v01 *= fast_sigmoid(v01);
      v10 *= fast_sigmoid(v10); v11 *= fast_sigmoid(v11);
    }
    C[(size_t)m0 * N + n0] = v00;
    C[(size_t)m0 * N + n1] = v01;
    C[(size_t)m1 * N + n0] = v10;
    C[(size_t)m1 * N + n1] = v11;
  }
}

// ---------------------------------------------------------------------------
// Fused additive-attention aspect layer. One block per (b,t) row.
// score_s = leaky( attn_w . tanh(q[b,t,:] + k[b,s,:]) + attn_b )
// out = layernorm( softmax(score) @ x  +  x )
// Never materializes the [B,T,T,D] tensor.
// ---------------------------------------------------------------------------
__global__ __launch_bounds__(256) void aspect_attn_k(
    const float* __restrict__ q, const float* __restrict__ k,
    const float* __restrict__ x, const float* __restrict__ attn_w,
    const float* __restrict__ attn_b, const float* __restrict__ ln_g,
    const float* __restrict__ ln_b, float* __restrict__ out)
{
  const int D = 256, T = 256;
  const int bt  = blockIdx.x;          // 0..1023
  const int b   = bt >> 8;
  const int tid = threadIdx.x;         // == d
  __shared__ float sq[256], sw[256], sc[256], red[256];
  sq[tid] = q[(size_t)bt * D + tid];
  sw[tid] = attn_w[tid];
  __syncthreads();

  const int lane = tid & 31, wv = tid >> 5;
  const float* kbase = k + (size_t)b * T * D;
  const float bconst = attn_b[0];
  for (int s = wv; s < T; s += 8) {
    const float* ks = kbase + (size_t)s * D;
    float p = 0.f;
    #pragma unroll
    for (int d = lane; d < D; d += 32)
      p += sw[d] * tanhf(sq[d] + ks[d]);
    for (int off = 16; off; off >>= 1) p += __shfl_xor(p, off, 32);
    float e = p + bconst;
    e = (e >= 0.f) ? e : 0.2f * e;     // LeakyReLU(0.2)
    if (lane == 0) sc[s] = e;
  }
  __syncthreads();

  // softmax over s
  float v = sc[tid];
  red[tid] = v; __syncthreads();
  for (int off = 128; off; off >>= 1) { if (tid < off) red[tid] = fmaxf(red[tid], red[tid + off]); __syncthreads(); }
  const float mx = red[0]; __syncthreads();
  float e = __expf(v - mx);
  red[tid] = e; __syncthreads();
  for (int off = 128; off; off >>= 1) { if (tid < off) red[tid] += red[tid + off]; __syncthreads(); }
  const float rdenom = fast_rcp(red[0]); __syncthreads();
  sc[tid] = e * rdenom;
  __syncthreads();

  // out_d = sum_s alpha_s x[b,s,d] + x[b,t,d]
  const float* xb = x + (size_t)b * T * D;
  float acc = 0.f;
  for (int s = 0; s < T; ++s) acc += sc[s] * xb[(size_t)s * D + tid];
  acc += x[(size_t)bt * D + tid];

  // layernorm
  red[tid] = acc; __syncthreads();
  for (int off = 128; off; off >>= 1) { if (tid < off) red[tid] += red[tid + off]; __syncthreads(); }
  const float mean = red[0] * (1.0f / 256.0f); __syncthreads();
  const float dv = acc - mean;
  red[tid] = dv * dv; __syncthreads();
  for (int off = 128; off; off >>= 1) { if (tid < off) red[tid] += red[tid + off]; __syncthreads(); }
  const float var = red[0] * (1.0f / 256.0f);
  out[(size_t)bt * D + tid] = dv * rsqrtf(var + 1e-5f) * ln_g[tid] + ln_b[tid];
}

// ---------------------------------------------------------------------------
// MHA core on packed qkv [B,T,3*sd]. One wave per (b,h,t). heads=4, T=256.
// ---------------------------------------------------------------------------
__global__ __launch_bounds__(256) void mha_attn_k(
    const float* __restrict__ qkv, float* __restrict__ o, int sd, int dh)
{
  const int T = 256;
  __shared__ float alpha[8][256];
  const int lane = threadIdx.x & 31;
  const int wv   = threadIdx.x >> 5;
  const int gw   = blockIdx.x * 8 + wv;     // (b*4 + h)*256 + t
  const int t  = gw & 255;
  const int bh = gw >> 8;
  const int h  = bh & 3;
  const int b  = bh >> 2;

  const size_t row3 = (size_t)3 * sd;
  const float* base = qkv + (size_t)b * T * row3;
  const float* qrow = base + (size_t)t * row3 + h * dh;
  const float* k0   = base + h * dh + sd;
  const float* v0   = base + h * dh + 2 * sd;
  const float rs = rsqrtf((float)dh);

  float loc[8];
  float mx = -3.4e38f;
  #pragma unroll
  for (int i = 0; i < 8; ++i) {
    const int s = lane + 32 * i;
    const float* kr = k0 + (size_t)s * row3;
    float p = 0.f;
    for (int d = 0; d < dh; ++d) p += qrow[d] * kr[d];
    p *= rs;
    loc[i] = p;
    mx = fmaxf(mx, p);
  }
  for (int off = 16; off; off >>= 1) mx = fmaxf(mx, __shfl_xor(mx, off, 32));
  float sum = 0.f;
  #pragma unroll
  for (int i = 0; i < 8; ++i) { loc[i] = __expf(loc[i] - mx); sum += loc[i]; }
  for (int off = 16; off; off >>= 1) sum += __shfl_xor(sum, off, 32);
  const float inv = fast_rcp(sum);
  #pragma unroll
  for (int i = 0; i < 8; ++i) alpha[wv][lane + 32 * i] = loc[i] * inv;
  __syncthreads();

  for (int d = lane; d < dh; d += 32) {
    float acc = 0.f;
    for (int s = 0; s < T; ++s) acc += alpha[wv][s] * v0[(size_t)s * row3 + d];
    o[((size_t)b * T + t) * sd + h * dh + d] = acc;
  }
}

// ---------------------------------------------------------------------------
// Depthwise conv K=5 (pad 2) + bias.  h:[B,T,C] -> out:[B,T,C]
// ---------------------------------------------------------------------------
__global__ void dwconv_k(const float* __restrict__ h, const float* __restrict__ w,
                         const float* __restrict__ bias, float* __restrict__ out, int C)
{
  const int idx = blockIdx.x * 256 + threadIdx.x;   // over 1024*C
  if (idx >= 1024 * C) return;
  const int c  = idx % C;
  const int bt = idx / C;
  const int t  = bt & 255;
  const int b  = bt >> 8;
  float acc = bias[c];
  #pragma unroll
  for (int j = 0; j < 5; ++j) {
    const int tt = t + j - 2;
    if (tt >= 0 && tt < 256)
      acc += h[((size_t)(b << 8) + tt) * C + c] * w[c * 5 + j];
  }
  out[idx] = acc;
}

// RMSNorm over rows of length n (<=256). One block per row.
__global__ __launch_bounds__(256) void rmsnorm_k(const float* __restrict__ x,
                                                 const float* __restrict__ g,
                                                 float* __restrict__ out, int n)
{
  __shared__ float red[256];
  const int row = blockIdx.x, tid = threadIdx.x;
  const float v = (tid < n) ? x[(size_t)row * n + tid] : 0.f;
  red[tid] = v * v; __syncthreads();
  for (int off = 128; off; off >>= 1) { if (tid < off) red[tid] += red[tid + off]; __syncthreads(); }
  const float r = rsqrtf(red[0] * fast_rcp((float)n) + 1e-6f);
  if (tid < n) out[(size_t)row * n + tid] = v * r * g[tid];
}

// GLU: y:[rows, 2m] -> out:[rows, m], out = a * sigmoid(g)
__global__ void glu_k(const float* __restrict__ y, float* __restrict__ out, int m, int n)
{
  const int idx = blockIdx.x * 256 + threadIdx.x;
  if (idx >= n) return;
  const int row = idx / m, j = idx % m;
  const float a = y[(size_t)row * 2 * m + j];
  const float g = y[(size_t)row * 2 * m + m + j];
  out[idx] = a * fast_sigmoid(g);
}

// h += softmax(attn_weight)[0]*hg + softmax(attn_weight)[1]*hl
__global__ void combine_k(float* __restrict__ h, const float* __restrict__ hg,
                          const float* __restrict__ hl, const float* __restrict__ aw, int n)
{
  const int idx = blockIdx.x * 256 + threadIdx.x;
  if (idx >= n) return;
  const float w0 = aw[0], w1 = aw[1];
  const float m = fmaxf(w0, w1);
  const float e0 = __expf(w0 - m), e1 = __expf(w1 - m);
  const float rs = fast_rcp(e0 + e1);
  h[idx] += (e0 * rs) * hg[idx] + (e1 * rs) * hl[idx];
}

__global__ void add_k(float* __restrict__ a, const float* __restrict__ b, int n)
{
  const int idx = blockIdx.x * 256 + threadIdx.x;
  if (idx < n) a[idx] += b[idx];
}

// fused += softmax(scale_weight)[i] * up
__global__ void accum_k(float* __restrict__ fused, const float* __restrict__ up,
                        const float* __restrict__ sw, int i, int n)
{
  const int idx = blockIdx.x * 256 + threadIdx.x;
  if (idx >= n) return;
  const float m = fmaxf(fmaxf(sw[0], sw[1]), sw[2]);
  const float e0 = __expf(sw[0] - m), e1 = __expf(sw[1] - m), e2 = __expf(sw[2] - m);
  const float w = ((i == 0) ? e0 : (i == 1) ? e1 : e2) * fast_rcp(e0 + e1 + e2);
  fused[idx] += w * up[idx];
}

__global__ void zero_k(float* __restrict__ p, int n)
{
  const int idx = blockIdx.x * 256 + threadIdx.x;
  if (idx < n) p[idx] = 0.0f;
}

// out = rmsnorm( x + sigmoid(res_gate)*fused, fuse_norm ).  Row length 256.
__global__ __launch_bounds__(256) void final_k(
    const float* __restrict__ x, const float* __restrict__ fused,
    const float* __restrict__ gate, const float* __restrict__ fn,
    float* __restrict__ out)
{
  __shared__ float red[256];
  const int row = blockIdx.x, tid = threadIdx.x;
  const float g = fast_sigmoid(gate[0]);
  const float v = x[(size_t)row * 256 + tid] + g * fused[(size_t)row * 256 + tid];
  red[tid] = v * v; __syncthreads();
  for (int off = 128; off; off >>= 1) { if (tid < off) red[tid] += red[tid + off]; __syncthreads(); }
  const float r = rsqrtf(red[0] * (1.0f / 256.0f) + 1e-6f);
  out[(size_t)row * 256 + tid] = v * r * fn[tid];
}

// ---------------------------------------------------------------------------
// Host orchestration
// ---------------------------------------------------------------------------
static inline void launch_gemm(const float* A, const float* W, const float* bias,
                               float* C, int M, int N, int K, int act,
                               hipStream_t s)
{
  const int tiles = (M / 32) * (N / 32);           // 32x32 supertiles
  const int blocks = (tiles + 7) / 8;
  gemm_wmma_f16<<<blocks, 256, 0, s>>>(A, W, bias, C, M, N, K, act);
}

extern "C" void kernel_launch(void* const* d_in, const int* in_sizes, int n_in,
                              void* d_out, int out_size, void* d_ws, size_t ws_size,
                              hipStream_t stream)
{
  (void)in_sizes; (void)n_in; (void)out_size; (void)ws_size;
  const int D = 256, T = 256, Bx = 4;
  const int M = Bx * T;                       // 1024 tokens

  const float* x_in = (const float*)d_in[0];

  float* ws = (float*)d_ws;
  float* xA    = ws; ws += M * D;
  float* xB    = ws; ws += M * D;
  float* qb    = ws; ws += M * D;
  float* kb    = ws; ws += M * D;
  float* hb    = ws; ws += M * D;
  float* hg    = ws; ws += M * D;
  float* fused = ws; ws += M * D;
  float* tA    = ws; ws += M * 1024;          // max 4*sd = 1024 cols
  float* tB    = ws; ws += M * 1024;

  // --- aspect layers (additive attention) ---
  const float* xcur = x_in;
  for (int l = 0; l < 2; ++l) {
    const int base = 1 + l * 8;
    const float* Wq_w   = (const float*)d_in[base + 0];
    const float* Wq_b   = (const float*)d_in[base + 1];
    const float* Wk_w   = (const float*)d_in[base + 2];
    const float* Wk_b   = (const float*)d_in[base + 3];
    const float* attn_w = (const float*)d_in[base + 4];
    const float* attn_b = (const float*)d_in[base + 5];
    const float* ln_g   = (const float*)d_in[base + 6];
    const float* ln_b   = (const float*)d_in[base + 7];

    launch_gemm(xcur, Wq_w, Wq_b, qb, M, D, D, 0, stream);
    launch_gemm(xcur, Wk_w, Wk_b, kb, M, D, D, 0, stream);
    float* xo = (l == 0) ? xA : xB;
    aspect_attn_k<<<M, 256, 0, stream>>>(qb, kb, xcur, attn_w, attn_b, ln_g, ln_b, xo);
    xcur = xo;
  }

  // --- GLMF ---
  const float* scale_weight = (const float*)d_in[68];
  const float* attn_weight  = (const float*)d_in[69];
  const float* res_gate     = (const float*)d_in[70];
  const float* fuse_norm    = (const float*)d_in[71];

  zero_k<<<(M * D + 255) / 256, 256, 0, stream>>>(fused, M * D);

  const int sds[3] = {256, 128, 64};
  for (int i = 0; i < 3; ++i) {
    const int bb = 17 + i * 17;
    const int sd = sds[i], dh = sd / 4;
    const float* down_w      = (const float*)d_in[bb + 0];
    const float* down_b      = (const float*)d_in[bb + 1];
    const float* in_proj_w   = (const float*)d_in[bb + 2];
    const float* in_proj_b   = (const float*)d_in[bb + 3];
    const float* out_proj_w  = (const float*)d_in[bb + 4];
    const float* out_proj_b  = (const float*)d_in[bb + 5];
    const float* cl_w        = (const float*)d_in[bb + 6];
    const float* cl_b        = (const float*)d_in[bb + 7];
    const float* cf_w        = (const float*)d_in[bb + 8];
    const float* cf_b        = (const float*)d_in[bb + 9];
    const float* ff_rms      = (const float*)d_in[bb + 10];
    const float* ff1_w       = (const float*)d_in[bb + 11];
    const float* ff1_b       = (const float*)d_in[bb + 12];
    const float* ff2_w       = (const float*)d_in[bb + 13];
    const float* ff2_b       = (const float*)d_in[bb + 14];
    const float* up_w        = (const float*)d_in[bb + 15];
    const float* up_b        = (const float*)d_in[bb + 16];

    // down projection
    launch_gemm(xcur, down_w, down_b, hb, M, sd, D, 0, stream);
    // qkv pack + MHA + out projection
    launch_gemm(hb, in_proj_w, in_proj_b, tA, M, 3 * sd, sd, 0, stream);
    mha_attn_k<<<(Bx * 4 * T) / 8, 256, 0, stream>>>(tA, tB, sd, dh);
    launch_gemm(tB, out_proj_w, out_proj_b, hg, M, sd, sd, 0, stream);
    // local branch: depthwise conv + pointwise(SiLU fused in GEMM epilogue)
    dwconv_k<<<(M * sd + 255) / 256, 256, 0, stream>>>(hb, cl_w, cl_b, tA, sd);
    launch_gemm(tA, cf_w, cf_b, tB, M, sd, sd, 1, stream);
    // gated combine
    combine_k<<<(M * sd + 255) / 256, 256, 0, stream>>>(hb, hg, tB, attn_weight, M * sd);
    // GLU feed-forward
    rmsnorm_k<<<M, 256, 0, stream>>>(hb, ff_rms, tA, sd);
    launch_gemm(tA, ff1_w, ff1_b, tB, M, 4 * sd, sd, 0, stream);
    glu_k<<<(M * 2 * sd + 255) / 256, 256, 0, stream>>>(tB, tA, 2 * sd, M * 2 * sd);
    launch_gemm(tA, ff2_w, ff2_b, tB, M, sd, 2 * sd, 0, stream);
    add_k<<<(M * sd + 255) / 256, 256, 0, stream>>>(hb, tB, M * sd);
    // up projection + weighted accumulation
    launch_gemm(hb, up_w, up_b, tA, M, D, sd, 0, stream);
    accum_k<<<(M * D + 255) / 256, 256, 0, stream>>>(fused, tA, scale_weight, i, M * D);
  }

  final_k<<<M, 256, 0, stream>>>(xcur, fused, res_gate, fuse_norm, (float*)d_out);
}
